// sLSTMCell_7224134992404
// MI455X (gfx1250) — matrix operand order repaired and checked by reading, not compiled, not run on hardware
//
#include <hip/hip_runtime.h>

// sLSTM persistent-scan kernel for gfx1250 (MI455X).
// grid = NH(8) workgroups, one head each; block = 512 threads (16 waves, wave32).
// Weights held in VGPRs as bf16 WMMA B-operands; mat-vec on the WMMA pipe;
// x prefetched with gfx1250 async global->LDS loads when available.

#define SEQn 8192
#define Hn   1024
#define NHn  8
#define DHn  128
#define NGn  4

typedef __attribute__((ext_vector_type(16))) __bf16 v16bf;
typedef __attribute__((ext_vector_type(8)))  __bf16 v8bf;
typedef __attribute__((ext_vector_type(8)))  float  v8f;

#if defined(__has_builtin)
#if __has_builtin(__builtin_amdgcn_global_load_async_to_lds_b32) && \
    __has_builtin(__builtin_amdgcn_s_wait_asynccnt)
#define USE_ASYNC 1
#endif
#endif
#ifndef USE_ASYNC
#define USE_ASYNC 0
#endif

#if USE_ASYNC
typedef __attribute__((address_space(1))) int* gptr_i32;
typedef __attribute__((address_space(3))) int* lptr_i32;
__device__ __forceinline__ void async_ld_b32(const float* g, float* l) {
    __builtin_amdgcn_global_load_async_to_lds_b32(
        (gptr_i32)(void*)const_cast<float*>(g),
        (lptr_i32)(void*)l, 0, 0);
}
#endif

__launch_bounds__(512, 1)
__global__ void slstm_scan_kernel(const float* __restrict__ x,     // [SEQ, NG*H]
                                  const float* __restrict__ Rk,    // [NH, DH, NG, DH]
                                  const float* __restrict__ bias,  // [NH, NG, DH]
                                  const float* __restrict__ st_in, // [4, H]
                                  float* __restrict__ out)         // [SEQ*H + 4*H]
{
    const int h    = blockIdx.x;
    const int tid  = threadIdx.x;
    const int wave = tid >> 5;
    const int lane = tid & 31;

    __shared__ __align__(16) __bf16 y_bf[DHn];      // y_prev (bf16) for A operand
    __shared__ float        ry[NGn * DHn];          // R @ y_prev (gate-major rows)
    __shared__ unsigned int nz_ball[4];             // per-wave ballots of (n != 0)
#if USE_ASYNC
    __shared__ float        xbuf[4][NGn * DHn];     // async x prefetch ring
#endif

    // ---------------- B operands: R^T tiles in bf16, resident in VGPRs ----
    // R_mat[row=(g*128+e)][d] = Rk[((h*128+d)*4+g)*128+e]
    // Tile T covers rows 16T..16T+15; wave w owns tiles 2w, 2w+1.
    // B layout (32x16 bf16, wave32): column n = lane%16; element s = K (8h+s | 16+8h+s-8), h = lane/16.
    v16bf Bt[2][4];
#pragma unroll
    for (int ct = 0; ct < 2; ++ct) {
        const int T = 2 * wave + ct;
#pragma unroll
        for (int c = 0; c < 4; ++c) {
            const int row = 16 * T + (lane & 15);
            const int g = row >> 7, e = row & 127;
            v16bf b;
#pragma unroll
            for (int n = 0; n < 16; ++n) {
                const int d = 32 * c + ((n < 8) ? 8 * (lane >> 4) + n : 16 + 8 * (lane >> 4) + (n - 8));
                const float* src = Rk + (size_t)(h * DHn + d) * (NGn * DHn);
                b[n] = (__bf16)src[g * DHn + e];
            }
            Bt[ct][c] = b;
        }
    }

    // ---------------- per-channel state in registers ----------------------
    float sy = 0.f, sc = 0.f, sn = 0.f, sm = 0.f;
    float b0 = 0.f, b1 = 0.f, b2 = 0.f, b3 = 0.f;
    if (tid < DHn) {
        sy = st_in[0 * Hn + h * DHn + tid];
        sc = st_in[1 * Hn + h * DHn + tid];
        sn = st_in[2 * Hn + h * DHn + tid];
        sm = st_in[3 * Hn + h * DHn + tid];
        b0 = bias[(h * NGn + 0) * DHn + tid];
        b1 = bias[(h * NGn + 1) * DHn + tid];
        b2 = bias[(h * NGn + 2) * DHn + tid];
        b3 = bias[(h * NGn + 3) * DHn + tid];
        y_bf[tid] = (__bf16)sy;
        // initial all(n==0) flag (exact per-head for zero-initialized state)
        unsigned long long bl = __ballot(sn != 0.0f);
        if (lane == 0) nz_ball[wave] = (unsigned int)bl;
    }

    // ---------------- x prefetch ------------------------------------------
#if USE_ASYNC
#pragma unroll
    for (int p = 0; p < 3; ++p) {
        const float* gp = x + (size_t)p * (NGn * Hn) + (tid >> 7) * Hn + h * DHn + (tid & 127);
        async_ld_b32(gp, &xbuf[p][tid]);
    }
#else
    float xr0 = 0.f, xr1 = 0.f, xr2 = 0.f, xr3 = 0.f;
    if (tid < DHn) {
        const float* gp = x + h * DHn + tid;
        xr0 = gp[0 * Hn]; xr1 = gp[1 * Hn]; xr2 = gp[2 * Hn]; xr3 = gp[3 * Hn];
    }
#endif

    // ---------------- sequential scan -------------------------------------
#pragma unroll 1
    for (int t = 0; t < SEQn; ++t) {
#if USE_ASYNC
        __builtin_amdgcn_s_wait_asynccnt(2);   // step-t x slot complete
#endif
        __syncthreads();                       // y_bf / xbuf[t%4] / nz_ball ready

        const bool nzero =
            ((nz_ball[0] | nz_ball[1] | nz_ball[2] | nz_ball[3]) == 0u);

#if USE_ASYNC
        if (t + 3 < SEQn) {                    // prefetch x[t+3] into free slot
            const float* gp = x + (size_t)(t + 3) * (NGn * Hn) + (tid >> 7) * Hn + h * DHn + (tid & 127);
            async_ld_b32(gp, &xbuf[(t + 3) & 3][tid]);
        }
#endif
        // A operand: y chunk broadcast to all 16 rows (16-bit A layout:
        // lanes<16 need K {0..7,16..23}; lanes>=16 need K {8..15,24..31}).
        const int off0 = (lane < 16) ? 0 : 8;
        v16bf A[4];
#pragma unroll
        for (int c = 0; c < 4; ++c) {
            v8bf lo = *(const v8bf*)&y_bf[32 * c + off0];
            v8bf hi = *(const v8bf*)&y_bf[32 * c + off0 + 16];
            A[c] = __builtin_shufflevector(lo, hi,
                    0, 1, 2, 3, 4, 5, 6, 7, 8, 9, 10, 11, 12, 13, 14, 15);
        }

        // D[i][j] = sum_c y_chunk_c . R_mat[16T+j][32c..] — row-replicated.
#pragma unroll
        for (int ct = 0; ct < 2; ++ct) {
            v8f acc = {};
#pragma unroll
            for (int c = 0; c < 4; ++c)
                acc = __builtin_amdgcn_wmma_f32_16x16x32_bf16(
                        false, A[c], false, Bt[ct][c], (short)0, acc, false, false);
            if (lane < 16)
                ry[(2 * wave + ct) * 16 + lane] = acc[0];   // M=0 row, N=lane
        }
        __syncthreads();                       // ry ready

        if (tid < DHn) {
#if USE_ASYNC
            const int slot = t & 3;
            const float iraw = xbuf[slot][0 * DHn + tid] + ry[0 * DHn + tid] + b0;
            const float fraw = xbuf[slot][1 * DHn + tid] + ry[1 * DHn + tid] + b1;
            const float zraw = xbuf[slot][2 * DHn + tid] + ry[2 * DHn + tid] + b2;
            const float oraw = xbuf[slot][3 * DHn + tid] + ry[3 * DHn + tid] + b3;
#else
            const float iraw = xr0 + ry[0 * DHn + tid] + b0;
            const float fraw = xr1 + ry[1 * DHn + tid] + b1;
            const float zraw = xr2 + ry[2 * DHn + tid] + b2;
            const float oraw = xr3 + ry[3 * DHn + tid] + b3;
#endif
            // stable log-sigmoid
            const float lsig = fminf(fraw, 0.f) - log1pf(__expf(-fabsf(fraw)));
            const float lfm  = sm + lsig;
            const float mnew = nzero ? iraw : fmaxf(iraw, lfm);
            const float og   = 1.f / (1.f + __expf(-oraw));
            const float ig   = fminf(__expf(iraw - mnew), 1.f);
            const float fg   = fminf(__expf(lfm - mnew), 1.f);
            const float cnew = fg * sc + ig * tanhf(zraw);
            const float nnew = fg * sn + ig;
            const float ynew = og * cnew / nnew;
            sc = cnew; sn = nnew; sm = mnew; sy = ynew;

            y_bf[tid] = (__bf16)ynew;                 // for next step's A
            out[(size_t)t * Hn + h * DHn + tid] = ynew;

            unsigned long long bl = __ballot(nnew != 0.0f);
            if (lane == 0) nz_ball[wave] = (unsigned int)bl;

#if !USE_ASYNC
            if (t + 1 < SEQn) {                       // register software pipeline
                const float* gp = x + (size_t)(t + 1) * (NGn * Hn) + h * DHn + tid;
                xr0 = gp[0 * Hn]; xr1 = gp[1 * Hn]; xr2 = gp[2 * Hn]; xr3 = gp[3 * Hn];
            }
#endif
        }
    }

    // ---------------- final states [y, c, n, m] ----------------------------
    if (tid < DHn) {
        float* fo = out + (size_t)SEQn * Hn;
        fo[0 * Hn + h * DHn + tid] = sy;
        fo[1 * Hn + h * DHn + tid] = sc;
        fo[2 * Hn + h * DHn + tid] = sn;
        fo[3 * Hn + h * DHn + tid] = sm;
    }
}

extern "C" void kernel_launch(void* const* d_in, const int* in_sizes, int n_in,
                              void* d_out, int out_size, void* d_ws, size_t ws_size,
                              hipStream_t stream) {
    (void)in_sizes; (void)n_in; (void)out_size; (void)d_ws; (void)ws_size;
    const float* x    = (const float*)d_in[0];
    const float* Rk   = (const float*)d_in[1];
    const float* bias = (const float*)d_in[2];
    const float* st   = (const float*)d_in[3];
    float* out = (float*)d_out;
    slstm_scan_kernel<<<NHn, 512, 0, stream>>>(x, Rk, bias, st, out);
}
